// MultiHeadAttention_6193342841469
// MI455X (gfx1250) — compile-verified, matrix operand order checked by reading, not run on hardware
//
#include <hip/hip_runtime.h>
#include <math.h>

// ---------------------------------------------------------------------------
// Types for CDNA5 WMMA (wave32, 16x16x32 bf16 -> f32)
// ---------------------------------------------------------------------------
typedef __attribute__((ext_vector_type(16))) __bf16 v16bf;
typedef __attribute__((ext_vector_type(8)))  float  v8f;

__device__ __forceinline__ v8f wmma_bf16(v16bf a, v16bf b, v8f c) {
  // (neg_a, A, neg_b, B, c_mod, C, reuse_a, reuse_b)
  return __builtin_amdgcn_wmma_f32_16x16x32_bf16(false, a, false, b, (short)0, c,
                                                 false, false);
}

// Load a 16x32 bf16 A/B fragment from LDS laid out as [row][k], row stride 32.
// Lanes 0-15 hold k=0..7 & 16..23, lanes 16-31 hold k=8..15 & 24..31
// (ISA 7.12.2 16-bit A layout; B uses the same with n=row).
__device__ __forceinline__ v16bf frag_lds(const __bf16* base, int row, int lane) {
  const int kb = (lane >> 4) * 8;
  const __bf16* p = base + row * 32;
  v16bf f;
  uint4* fp = reinterpret_cast<uint4*>(&f);
  fp[0] = *reinterpret_cast<const uint4*>(p + kb);        // elems 0..7  -> k=kb..kb+7
  fp[1] = *reinterpret_cast<const uint4*>(p + kb + 16);   // elems 8..15 -> k=kb+16..kb+23
  return f;
}

// Build a fragment straight from a global f32 row (contiguous in k).
__device__ __forceinline__ v16bf frag_f32row(const float* rowptr, int lane) {
  const int kb = (lane >> 4) * 8;
  v16bf f;
#pragma unroll
  for (int j = 0; j < 8; ++j) f[j]     = (__bf16)rowptr[kb + j];
#pragma unroll
  for (int j = 0; j < 8; ++j) f[8 + j] = (__bf16)rowptr[kb + 16 + j];
  return f;
}

// ---------------------------------------------------------------------------
// Kernel 1: generic strided-batched GEMM, f32 in -> bf16 WMMA -> f32 out.
// Block: 256 threads = 8 waves; tile 128x64; wave (wm 0..3, wn 0..1) computes
// 32 rows x 32 cols: 2 A-frags x 2 B-frags -> 4 WMMAs per 32-deep k-step.
// Batch index z decomposes as b = z/nH, h = z%nH with independent strides.
// Optional residual added at epilogue (same layout as C).
// ---------------------------------------------------------------------------
__global__ __launch_bounds__(256)
void gemm_bf16_wmma(const float* __restrict__ A, const float* __restrict__ B,
                    float* __restrict__ C, const float* __restrict__ Rres,
                    int K, int lda, int ldb, int ldc, int nH,
                    long long sAb, long long sAh,
                    long long sBb, long long sBh,
                    long long sCb, long long sCh) {
  __shared__ __align__(16) __bf16 As[128 * 32];  // [m][k]
  __shared__ __align__(16) __bf16 Bs[64 * 32];   // [n][k] (transposed at load)

  const int tid  = threadIdx.x;
  const int lane = tid & 31;
  const int wave = tid >> 5;
  const int wm   = wave >> 1;   // 0..3  -> 32-row slice
  const int wn   = wave & 1;    // 0..1  -> 32-col slice

  const int z = blockIdx.z;
  const int b = z / nH;
  const int h = z - b * nH;
  const long long offA = (long long)b * sAb + (long long)h * sAh;
  const long long offB = (long long)b * sBb + (long long)h * sBh;
  const long long offC = (long long)b * sCb + (long long)h * sCh;

  const int m0 = blockIdx.y * 128;
  const int n0 = blockIdx.x * 64;

  v8f acc00 = {}, acc01 = {}, acc10 = {}, acc11 = {};

  // staging coordinates (hoisted out of the k loop)
  const int ar = tid >> 1;            // 0..127  A row
  const int ac = (tid & 1) * 16;      // 0 / 16  A col base (16 floats)
  const int bk = (tid & 15) * 2;      // k pair  0,2,..,30
  const int bc = (tid >> 4) * 4;      // 4-col group 0,4,..,60

  for (int k0 = 0; k0 < K; k0 += 32) {
    // --- stage A tile 128x32 (each thread: 16 contiguous floats -> 16 bf16)
    {
      const float* src = A + offA + (long long)(m0 + ar) * lda + k0 + ac;
      float4 a0 = *reinterpret_cast<const float4*>(src);
      float4 a1 = *reinterpret_cast<const float4*>(src + 4);
      float4 a2 = *reinterpret_cast<const float4*>(src + 8);
      float4 a3 = *reinterpret_cast<const float4*>(src + 12);
      if (k0 + 32 < K) __builtin_prefetch(src + 32, 0, 1);
      __bf16* d = &As[ar * 32 + ac];
      d[0]  = (__bf16)a0.x; d[1]  = (__bf16)a0.y; d[2]  = (__bf16)a0.z; d[3]  = (__bf16)a0.w;
      d[4]  = (__bf16)a1.x; d[5]  = (__bf16)a1.y; d[6]  = (__bf16)a1.z; d[7]  = (__bf16)a1.w;
      d[8]  = (__bf16)a2.x; d[9]  = (__bf16)a2.y; d[10] = (__bf16)a2.z; d[11] = (__bf16)a2.w;
      d[12] = (__bf16)a3.x; d[13] = (__bf16)a3.y; d[14] = (__bf16)a3.z; d[15] = (__bf16)a3.w;
    }
    // --- stage B tile 32x64 transposed into [n][k]; adjacent-k pairs per
    //     column so each pair merges into a ds_store_b32.
    {
      const float* src0 = B + offB + (long long)(k0 + bk) * ldb + n0 + bc;
      const float* src1 = src0 + ldb;
      float4 r0 = *reinterpret_cast<const float4*>(src0);
      float4 r1 = *reinterpret_cast<const float4*>(src1);
      if (k0 + 32 < K) __builtin_prefetch(src0 + 32LL * ldb, 0, 1);
      __bf16* d = &Bs[bc * 32 + bk];
      d[0]          = (__bf16)r0.x; d[1]          = (__bf16)r1.x;
      d[32]         = (__bf16)r0.y; d[33]         = (__bf16)r1.y;
      d[64]         = (__bf16)r0.z; d[65]         = (__bf16)r1.z;
      d[96]         = (__bf16)r0.w; d[97]         = (__bf16)r1.w;
    }
    __syncthreads();

    const v16bf fa0 = frag_lds(As, wm * 32 + (lane & 15), lane);
    const v16bf fa1 = frag_lds(As, wm * 32 + 16 + (lane & 15), lane);
    const v16bf fb0 = frag_lds(Bs, wn * 32 + (lane & 15), lane);
    const v16bf fb1 = frag_lds(Bs, wn * 32 + 16 + (lane & 15), lane);
    acc00 = wmma_bf16(fa0, fb0, acc00);
    acc01 = wmma_bf16(fa0, fb1, acc01);
    acc10 = wmma_bf16(fa1, fb0, acc10);
    acc11 = wmma_bf16(fa1, fb1, acc11);

    __syncthreads();
  }

  // --- epilogue: lane holds col n = lane&15, rows (lane>>4)*8 + r
  const int rbase = (lane >> 4) * 8;
  const int col   = n0 + wn * 32 + (lane & 15);
#pragma unroll
  for (int r = 0; r < 8; ++r) {
    const long long row0 = m0 + wm * 32 + rbase + r;        // acc0x tiles
    const long long row1 = row0 + 16;                       // acc1x tiles
    const long long i00  = offC + row0 * ldc + col;
    const long long i10  = offC + row1 * ldc + col;
    float v00 = acc00[r], v01 = acc01[r];
    float v10 = acc10[r], v11 = acc11[r];
    if (Rres) {
      v00 += Rres[i00]; v01 += Rres[i00 + 16];
      v10 += Rres[i10]; v11 += Rres[i10 + 16];
    }
    C[i00]      = v00;
    C[i00 + 16] = v01;
    C[i10]      = v10;
    C[i10 + 16] = v11;
  }
}

// ---------------------------------------------------------------------------
// Kernel 2: scores pass. One wave per (b, h, 16-row q tile).
// score = (Q K^T)/8 + conv_w*lps + conv_b, masked; raw scores written to the
// attn output region; online row max / sumexp kept in registers.
// Q/K fragments come straight from global (d=64 contiguous per row).
// ---------------------------------------------------------------------------
__global__ __launch_bounds__(32)
void attn_scores_pass1(const float* __restrict__ Qp, const float* __restrict__ Kp,
                       const float* __restrict__ lps, const unsigned char* __restrict__ mask,
                       const float* __restrict__ convw, const float* __restrict__ convb,
                       float* __restrict__ attn, float* __restrict__ rowmax,
                       float* __restrict__ rowsum) {
  const int S = 1024, H = 16, DM = 1024, D = 64;
  const int lane = threadIdx.x;
  const int qt = blockIdx.x;   // 0..63
  const int h  = blockIdx.y;   // 0..15
  const int b  = blockIdx.z;   // 0..3

  const float cw = convw[0];
  const float cb = convb[0];

  const float* Qb = Qp + (long long)b * S * DM + h * D;
  const float* Kb = Kp + (long long)b * S * DM + h * D;

  const int m    = lane & 15;
  const int half = lane >> 4;

  // A fragment: Q rows qt*16+m, d = 0..31 and 32..63
  const float* qrow = Qb + (long long)(qt * 16 + m) * DM;
  const v16bf aq0 = frag_f32row(qrow, lane);
  const v16bf aq1 = frag_f32row(qrow + 32, lane);

  float rmax[8], rsum[8];
#pragma unroll
  for (int r = 0; r < 8; ++r) { rmax[r] = -__builtin_inff(); rsum[r] = 0.f; }

  const long long attnBase = (((long long)b * H + h) * S + qt * 16) * S;

  for (int kt = 0; kt < S / 16; ++kt) {
    // B fragment: lane holds column n = lane&15 -> K row kt*16+n, elems over d
    const float* krow = Kb + (long long)(kt * 16 + m) * DM;
    v16bf bk0 = frag_f32row(krow, lane);
    v16bf bk1 = frag_f32row(krow + 32, lane);
    v8f c = {};
    c = wmma_bf16(aq0, bk0, c);
    c = wmma_bf16(aq1, bk1, c);

#pragma unroll
    for (int r = 0; r < 8; ++r) {
      const int rowt = half * 8 + r;           // row within q tile
      const int qq   = qt * 16 + rowt;
      const int colk = kt * 16 + (lane & 15);
      float s = c[r] * 0.125f + cw * lps[(long long)qq * S + colk] + cb;
      if (mask[((long long)b * S + qq) * S + colk]) s = -1.0e9f;
      attn[attnBase + (long long)rowt * S + colk] = s;   // raw score

      // row reductions across the 16 lanes of this half
      float tm = s;
#pragma unroll
      for (int off = 1; off < 16; off <<= 1) tm = fmaxf(tm, __shfl_xor(tm, off, 32));
      const float nm = fmaxf(rmax[r], tm);
      float te = __expf(s - nm);
#pragma unroll
      for (int off = 1; off < 16; off <<= 1) te += __shfl_xor(te, off, 32);
      rsum[r] = rsum[r] * __expf(rmax[r] - nm) + te;
      rmax[r] = nm;
    }
  }

  if ((lane & 15) == 0) {
    const long long base = ((long long)b * H + h) * S + qt * 16 + half * 8;
#pragma unroll
    for (int r = 0; r < 8; ++r) {
      rowmax[base + r] = rmax[r];
      rowsum[base + r] = rsum[r];
    }
  }
}

// ---------------------------------------------------------------------------
// Kernel 3: normalize raw scores in place: attn = exp(s - max) / sum
// ---------------------------------------------------------------------------
__global__ __launch_bounds__(256)
void attn_normalize(float* __restrict__ attn, const float* __restrict__ rowmax,
                    const float* __restrict__ rowsum) {
  const long long i = (long long)blockIdx.x * blockDim.x + threadIdx.x;
  const long long row = i >> 10;  // / S
  attn[i] = __expf(attn[i] - rowmax[row]) * (1.0f / rowsum[row]);
}

// ---------------------------------------------------------------------------
// Kernel 4: LayerNorm over rows of 1024, one 256-thread block per row.
// ---------------------------------------------------------------------------
__global__ __launch_bounds__(256)
void layernorm_k(const float* __restrict__ x, float* __restrict__ out) {
  const int D = 1024;
  __shared__ float sbuf[16];
  const long long row = blockIdx.x;
  const float* xr = x + row * D;

  float s = 0.f, s2 = 0.f;
  for (int i = threadIdx.x; i < D; i += 256) {
    float v = xr[i];
    s += v; s2 += v * v;
  }
#pragma unroll
  for (int off = 16; off > 0; off >>= 1) {
    s  += __shfl_xor(s,  off, 32);
    s2 += __shfl_xor(s2, off, 32);
  }
  const int wave = threadIdx.x >> 5;
  if ((threadIdx.x & 31) == 0) { sbuf[wave] = s; sbuf[8 + wave] = s2; }
  __syncthreads();
  if (threadIdx.x == 0) {
    float ts = 0.f, ts2 = 0.f;
#pragma unroll
    for (int w = 0; w < 8; ++w) { ts += sbuf[w]; ts2 += sbuf[8 + w]; }
    const float mu = ts / (float)D;
    sbuf[0] = mu;
    sbuf[1] = rsqrtf(ts2 / (float)D - mu * mu + 1e-5f);
  }
  __syncthreads();
  const float mu  = sbuf[0];
  const float inv = sbuf[1];
  for (int i = threadIdx.x; i < D; i += 256) out[row * D + i] = (xr[i] - mu) * inv;
}

// ---------------------------------------------------------------------------
// Host launch
// ---------------------------------------------------------------------------
extern "C" void kernel_launch(void* const* d_in, const int* in_sizes, int n_in,
                              void* d_out, int out_size, void* d_ws, size_t ws_size,
                              hipStream_t stream) {
  const int B = 4, S = 1024, DM = 1024, H = 16;
  const long long SD = (long long)S * DM;          // 1M elements

  const float* input_Q = (const float*)d_in[0];
  const float* input_K = (const float*)d_in[1];
  const float* input_V = (const float*)d_in[2];
  const unsigned char* attn_mask = (const unsigned char*)d_in[3];
  const float* lps  = (const float*)d_in[4];
  const float* W_Q  = (const float*)d_in[5];
  const float* W_K  = (const float*)d_in[6];
  const float* W_V  = (const float*)d_in[7];
  const float* W_fc = (const float*)d_in[8];
  const float* conv_w = (const float*)d_in[9];
  const float* conv_b = (const float*)d_in[10];

  float* out  = (float*)d_out;                 // [4,1024,1024]
  float* attn = out + (long long)B * SD;       // [4,16,1024,1024]

  float* ws  = (float*)d_ws;
  float* Qp  = ws;                             // [B*S, 1024] f32
  float* Kp  = ws + 1 * (long long)B * SD;
  float* Vp  = ws + 2 * (long long)B * SD;
  float* ctx = ws + 3 * (long long)B * SD;
  float* rowmax  = ws + 4 * (long long)B * SD;          // B*H*S
  float* rowsum  = rowmax + (long long)B * H * S;
  float* out_pre = Kp;                          // Kp dead after score pass

  dim3 blk(256);

  // 1) projections: [4096,1024] = X @ W   (batch = 1), tile 128x64
  dim3 gproj(DM / 64, (B * S) / 128, 1);
  gemm_bf16_wmma<<<gproj, blk, 0, stream>>>(input_Q, W_Q, Qp, nullptr,
      DM, DM, DM, DM, 1, 0, 0, 0, 0, 0, 0);
  gemm_bf16_wmma<<<gproj, blk, 0, stream>>>(input_K, W_K, Kp, nullptr,
      DM, DM, DM, DM, 1, 0, 0, 0, 0, 0, 0);
  gemm_bf16_wmma<<<gproj, blk, 0, stream>>>(input_V, W_V, Vp, nullptr,
      DM, DM, DM, DM, 1, 0, 0, 0, 0, 0, 0);

  // 2) scores + bias + mask + online softmax stats -> raw scores in attn
  dim3 gsc(S / 16, H, B);
  attn_scores_pass1<<<gsc, dim3(32), 0, stream>>>(Qp, Kp, lps, attn_mask,
      conv_w, conv_b, attn, rowmax, rowsum);

  // 3) normalize attn in place
  const long long nattn = (long long)B * H * S * S;
  attn_normalize<<<dim3((unsigned)(nattn / 256)), blk, 0, stream>>>(attn, rowmax, rowsum);

  // 4) context = attn @ V, batched over (b,h): M=1024, N=64, K=1024
  dim3 gctx(1, S / 128, B * H);
  gemm_bf16_wmma<<<gctx, blk, 0, stream>>>(attn, Vp, ctx, nullptr,
      S, S, DM, DM, H,
      (long long)H * S * S, (long long)S * S,   // A strides (b, h)
      SD, 64,                                   // B strides
      SD, 64);                                  // C strides

  // 5) out_pre = ctx @ W_fc + input_Q
  gemm_bf16_wmma<<<gproj, blk, 0, stream>>>(ctx, W_fc, out_pre, input_Q,
      DM, DM, DM, DM, 1, 0, 0, 0, 0, 0, 0);

  // 6) LayerNorm rows -> out
  layernorm_k<<<dim3(B * S), blk, 0, stream>>>(out_pre, out);
}
